// sLSTMBlock_58067957842088
// MI455X (gfx1250) — matrix operand order repaired and checked by reading, not compiled, not run on hardware
//
#include <hip/hip_runtime.h>
#include <hip/hip_bf16.h>

typedef __attribute__((ext_vector_type(16))) __bf16        v16bf;
typedef __attribute__((ext_vector_type(8)))  float         v8f;
typedef __attribute__((ext_vector_type(8)))  unsigned int  v8u;

#define H128    128
#define NWAVES  16                 // 512 threads/block, one 256KB B image in LDS per block
#define BFRAG_DWORDS 65536         // 8 n * 8 kt * 4 gates * 256 dwords = 256 KB

// sched_group_barrier masks
#define SGB_MFMA    0x008          // matches WMMA on gfx12xx
#define SGB_DSREAD  0x100

// round-to-nearest-even f32 -> bf16, pack two into a dword (lo = first)
__device__ __forceinline__ unsigned int bfpack(float a, float b) {
    unsigned int ua = __float_as_uint(a);
    unsigned int ub = __float_as_uint(b);
    ua += 0x7fffu + ((ua >> 16) & 1u);
    ub += 0x7fffu + ((ub >> 16) & 1u);
    return (ua >> 16) | (ub & 0xffff0000u);
}

__device__ __forceinline__ float sigmoidf(float x) {
    return __fdividef(1.0f, 1.0f + __expf(-x));
}

// ---------------------------------------------------------------------------
// Prep: pack [W_g | U_g]^T (K=256, N=128) into bf16 WMMA B-fragments in d_ws.
// Fragment order: frag = ((n*8 + kt)*4 + g)  -> the 4 gate fragments of one
// (n,kt) are contiguous (4 KB group, addressable via DS immediate offsets).
// Within a fragment (256 dwords), dword (l,v) holds the bf16 pair for
// K = kt*32 + (l>=16 ? 8 : 0) + 16*(v>=4) + 2*(v&3), column N = n*16 + (l&15).
// ---------------------------------------------------------------------------
__global__ __launch_bounds__(256) void slstm_prep_b(
    const float* __restrict__ Wi, const float* __restrict__ Ui,
    const float* __restrict__ Wf, const float* __restrict__ Uf,
    const float* __restrict__ Wo, const float* __restrict__ Uo,
    const float* __restrict__ Wz, const float* __restrict__ Uz,
    unsigned int* __restrict__ ws)
{
    const int frag = blockIdx.x;          // 0..255
    const int g  = frag & 3;
    const int kt = (frag >> 2) & 7;
    const int n  = frag >> 5;
    const int tid = threadIdx.x;          // 0..255
    const int l = tid >> 3;               // consumer lane 0..31
    const int v = tid & 7;                // consumer VGPR 0..7

    const float* W; const float* U;
    if      (g == 0) { W = Wi; U = Ui; }
    else if (g == 1) { W = Wf; U = Uf; }
    else if (g == 2) { W = Wo; U = Uo; }
    else             { W = Wz; U = Uz; }

    const int ncol = n * 16 + (l & 15);
    const int kg   = kt * 32 + ((l >= 16) ? 8 : 0) + ((v & 4) << 2) + ((v & 3) << 1);

    float e0, e1;
    if (kg < 128) { e0 = W[ncol * 128 + kg];         e1 = W[ncol * 128 + kg + 1]; }
    else          { e0 = U[ncol * 128 + (kg - 128)]; e1 = U[ncol * 128 + (kg - 127)]; }

    ws[frag * 256 + l * 8 + v] = bfpack(e0, e1);
}

// ---------------------------------------------------------------------------
// Main: B fragments staged once into LDS per block; each wave handles 32 batch
// rows (two 16-row M-tiles) per iteration. Direct-global A fragments; LDS B
// reads pipelined in half-groups (4 DS loads / 4 WMMAs, depth 2) via a
// rotating 3-slot buffer + sched_group_barrier.
// ---------------------------------------------------------------------------
__global__ __launch_bounds__(NWAVES * 32) void slstm_main(
    const float* __restrict__ x,      const float* __restrict__ h_prev,
    const float* __restrict__ c_prev, const float* __restrict__ n_prev,
    const float* __restrict__ b_i,    const float* __restrict__ b_f,
    const float* __restrict__ b_o,    const float* __restrict__ b_z,
    const unsigned int* __restrict__ wsb,
    float* __restrict__ out_h, float* __restrict__ out_c, float* __restrict__ out_n,
    int nrows)
{
    __shared__ unsigned int bsh[BFRAG_DWORDS];      // 256 KB of bf16 B fragments

    // ---- stage B fragments global -> LDS once per block ----
    {
        const uint4* src = (const uint4*)wsb;
        uint4*       dst = (uint4*)bsh;
        for (int i = threadIdx.x; i < (BFRAG_DWORDS >> 2); i += (NWAVES * 32))
            dst[i] = src[i];
    }
    __syncthreads();

    const int lane = threadIdx.x & 31;
    const int wv   = threadIdx.x >> 5;

    const int pairs      = nrows >> 5;                      // 32 rows per pair
    const int wave_slot  = blockIdx.x * NWAVES + wv;
    const int wave_count = gridDim.x * NWAVES;

    const unsigned int row  = lane & 15;
    const unsigned int lhi8 = (lane >> 4) << 3;

    const v8u* bsh8 = (const v8u*)bsh;

    for (int p = wave_slot; p < pairs; p += wave_count) {
        const unsigned int m0 = (unsigned int)p * 32u;

        // ---- build bf16 A fragments directly from global (L0-resident slab) ----
        // A layout (16-bit, 16x32): M = lane&15; K = kt*32 + lhi8 + 16*(v>=4) + 2*(v&3)
        v8u afrag[2][8];
        #pragma unroll
        for (int s = 0; s < 2; ++s) {
            const float* xb = x      + (size_t)(m0 + 16u * s) * H128 + row * H128;
            const float* hb = h_prev + (size_t)(m0 + 16u * s) * H128 + row * H128;
            #pragma unroll
            for (int kt = 0; kt < 8; ++kt) {
                const float* base = (kt < 4) ? xb : hb;     // K<128 from x, else h_prev
                const unsigned int k0 = (unsigned int)(kt & 3) * 32u + lhi8;
                #pragma unroll
                for (int v = 0; v < 8; ++v) {
                    const unsigned int k = k0 + ((v & 4) << 2) + ((v & 3) << 1);
                    float2 t = *(const float2*)(base + k);
                    afrag[s][kt][v] = bfpack(t.x, t.y);
                }
            }
        }

        // ---- GEMM (half-group pipelined LDS B) + fused epilogue, 8 N-tiles ----
        for (int n = 0; n < 8; ++n) {
            const unsigned int col = (unsigned int)n * 16u + (lane & 15);
            const float bia = b_i[col], bif = b_f[col], bio = b_o[col], biz = b_z[col];

            // one base for the whole n-tile: kt*4096B + g*1024B fit DS imm offset
            const v8u* bbase = bsh8 + ((unsigned)n * 1024u + (unsigned)lane);

            v8f acc[2][4];
            #pragma unroll
            for (int s = 0; s < 2; ++s)
                #pragma unroll
                for (int g = 0; g < 4; ++g)
                    acc[s][g] = (v8f)0.0f;

            // 16 half-groups: hg -> (kt = hg>>1, gates 2*(hg&1), 2*(hg&1)+1)
            // rotating 3-slot buffer, lookahead 2 half-groups (== 8 WMMAs)
            v8u buf[3][2];
            #pragma unroll
            for (int hg = 0; hg < 2; ++hg) {
                const unsigned int off = (unsigned)(hg >> 1) * 128u + (unsigned)(hg & 1) * 64u;
                buf[hg][0] = bbase[off];
                buf[hg][1] = bbase[off + 32u];
                __builtin_amdgcn_sched_group_barrier(SGB_DSREAD, 4, 0);
            }

            #pragma unroll
            for (int hg = 0; hg < 16; ++hg) {
                if (hg < 14) {
                    const int ld = hg + 2;
                    const unsigned int off = (unsigned)(ld >> 1) * 128u + (unsigned)(ld & 1) * 64u;
                    buf[ld % 3][0] = bbase[off];
                    buf[ld % 3][1] = bbase[off + 32u];
                }
                const int kt = hg >> 1;
                const int gb = (hg & 1) * 2;
                #pragma unroll
                for (int gg = 0; gg < 2; ++gg) {
                    #pragma unroll
                    for (int s = 0; s < 2; ++s) {
                        acc[s][gb + gg] = __builtin_amdgcn_wmma_f32_16x16x32_bf16(
                            false, __builtin_bit_cast(v16bf, afrag[s][kt]),
                            false, __builtin_bit_cast(v16bf, buf[hg % 3][gg]),
                            (short)0, acc[s][gb + gg], false, false);
                    }
                }
                if (hg < 14)
                    __builtin_amdgcn_sched_group_barrier(SGB_DSREAD, 4, 0);
                __builtin_amdgcn_sched_group_barrier(SGB_MFMA, 4, 0);
            }

            // ---- pointwise sLSTM epilogue ----
            #pragma unroll
            for (int s = 0; s < 2; ++s) {
                const unsigned int mrow = m0 + 16u * s + lhi8;          // + v below
                const unsigned int base = mrow * (unsigned)H128 + col;  // fits 32-bit
                const float* pc = c_prev + base;
                const float* pn = n_prev + base;
                float* ph  = out_h + base;
                float* pcc = out_c + base;
                float* pnn = out_n + base;
                #pragma unroll
                for (int v = 0; v < 8; ++v) {
                    const float cp = pc[v * H128];
                    const float np = pn[v * H128];

                    const float iv = acc[s][0][v] + bia;
                    const float fv = acc[s][1][v] + bif;
                    const float ov = acc[s][2][v] + bio;
                    const float zv = acc[s][3][v] + biz;

                    const float mt = fmaxf(fv, iv);
                    const float it = __expf(iv - mt);
                    const float ft = sigmoidf(fv) + __expf(fv - mt);
                    const float ot = sigmoidf(ov);
                    const float zt = tanhf(zv);

                    const float ct = ft * cp + it * zt;
                    const float nt = ft * np + it;
                    const float ht = ot * __fdividef(ct, nt + 1e-8f);

                    ph[v * H128]  = ht;
                    pcc[v * H128] = ct;
                    pnn[v * H128] = nt;
                }
            }
        }
    }
}

extern "C" void kernel_launch(void* const* d_in, const int* in_sizes, int n_in,
                              void* d_out, int out_size, void* d_ws, size_t ws_size,
                              hipStream_t stream) {
    const float* x      = (const float*)d_in[0];
    const float* h_prev = (const float*)d_in[1];
    const float* c_prev = (const float*)d_in[2];
    const float* n_prev = (const float*)d_in[3];
    const float* W_i = (const float*)d_in[4];
    const float* U_i = (const float*)d_in[5];
    const float* W_f = (const float*)d_in[6];
    const float* U_f = (const float*)d_in[7];
    const float* W_o = (const float*)d_in[8];
    const float* U_o = (const float*)d_in[9];
    const float* W_z = (const float*)d_in[10];
    const float* U_z = (const float*)d_in[11];
    const float* b_i = (const float*)d_in[12];
    const float* b_f = (const float*)d_in[13];
    const float* b_o = (const float*)d_in[14];
    const float* b_z = (const float*)d_in[15];

    const int nrows = in_sizes[0] / H128;            // B
    unsigned int* wsb = (unsigned int*)d_ws;         // 256 KB of B fragments

    float* out_h = (float*)d_out;
    float* out_c = out_h + (size_t)nrows * H128;
    float* out_n = out_c + (size_t)nrows * H128;

    // 1) pack weights into WMMA B fragments (bf16)
    slstm_prep_b<<<256, 256, 0, stream>>>(W_i, U_i, W_f, U_f, W_o, U_o, W_z, U_z, wsb);

    // 2) fused gates + pointwise
    const int pairs = nrows >> 5;
    int blocks = (pairs + NWAVES - 1) / NWAVES;
    if (blocks > 8192) blocks = 8192;
    if (blocks < 1) blocks = 1;
    slstm_main<<<blocks, NWAVES * 32, 0, stream>>>(
        x, h_prev, c_prev, n_prev, b_i, b_f, b_o, b_z,
        wsb, out_h, out_c, out_n, nrows);
}
